// SVEBackbone_81698867904874
// MI455X (gfx1250) — matrix operand order repaired and checked by reading, not compiled, not run on hardware
//
#include <hip/hip_runtime.h>
#include <hip/hip_bf16.h>

// ---------------------------------------------------------------------------
// CDNA5 WMMA types (wave32): v_wmma_f32_16x16x32_bf16
// ---------------------------------------------------------------------------
typedef __attribute__((ext_vector_type(16))) __bf16 v16bf;
typedef __attribute__((ext_vector_type(8)))  __bf16 v8bf;
typedef __attribute__((ext_vector_type(8)))  float  v8f;

#define DEVFN __device__ __forceinline__

DEVFN __bf16 f2bf(float f) {
  unsigned u = __builtin_bit_cast(unsigned, f);
  u = (u + 0x7FFFu + ((u >> 16) & 1u)) >> 16;   // round-to-nearest-even
  return __builtin_bit_cast(__bf16, (unsigned short)u);
}

// searchsorted(left) + clip + equality check (matches reference _lookup)
DEVFN int bsearch_key(const int* __restrict__ keys, int n, int q) {
  int lo = 0, hi = n;
  while (lo < hi) { int mid = (lo + hi) >> 1; if (keys[mid] < q) lo = mid + 1; else hi = mid; }
  int i = lo < n ? lo : n - 1;
  return (keys[i] == q) ? i : -1;
}

// ---------------------------------------------------------------------------
// Utility kernels
// ---------------------------------------------------------------------------
__global__ void k_fill(float* __restrict__ b, long n) {
  long t = (long)blockIdx.x * blockDim.x + threadIdx.x;
  if (t < n) b[t] = 0.0f;
}

__global__ void k_keys(const int* __restrict__ coords, int N,
                       int d0, int d1, int d2, int* __restrict__ keys) {
  int i = blockIdx.x * blockDim.x + threadIdx.x;
  if (i >= N) return;
  const int* c = coords + i * 4;
  keys[i] = ((c[0] * d0 + c[1]) * d1 + c[2]) * d2 + c[3];
}

__global__ void k_scatter_add(const float* __restrict__ f, const int* __restrict__ coords,
                              int Nsrc, int Cf, const int* __restrict__ keys0, int N0,
                              int d0, int d1, int d2, float* __restrict__ f0) {
  int t = blockIdx.x * blockDim.x + threadIdx.x;
  if (t >= Nsrc * Cf) return;
  int r = t / Cf, c = t % Cf;
  const int* cc = coords + r * 4;
  int q = ((cc[0] * d0 + cc[1]) * d1 + cc[2]) * d2 + cc[3];
  int j = bsearch_key(keys0, N0, q);
  if (j >= 0) atomicAdd(&f0[(long)j * Cf + c], f[t]);
}

// gather-index build: forward sparse conv   i = o*s - p + k
__global__ void k_gidx_conv(const int* __restrict__ oc, int Nout,
                            const int* __restrict__ ikeys, int Nin,
                            int kd, int kh, int kw, int sz, int sy, int sx,
                            int pz, int py, int px, int d0, int d1, int d2,
                            int* __restrict__ gidx) {
  long t = (long)blockIdx.x * blockDim.x + threadIdx.x;
  int K = kd * kh * kw;
  if (t >= (long)K * Nout) return;
  int ki = (int)(t / Nout), o = (int)(t % Nout);
  int kz = ki / (kh * kw), ky = (ki / kw) % kh, kx = ki % kw;
  const int* c = oc + o * 4;
  int z = c[1] * sz - pz + kz;
  int y = c[2] * sy - py + ky;
  int x = c[3] * sx - px + kx;
  int idx = -1;
  if (z >= 0 && z < d0 && y >= 0 && y < d1 && x >= 0 && x < d2) {
    int q = ((c[0] * d0 + z) * d1 + y) * d2 + x;
    idx = bsearch_key(ikeys, Nin, q);
  }
  gidx[t] = idx;
}

// gather-index build: inverse conv   j = (o + p - k)/s when divisible
__global__ void k_gidx_inv(const int* __restrict__ oc, int Nout,
                           const int* __restrict__ ikeys, int Nin,
                           int kd, int kh, int kw, int sz, int sy, int sx,
                           int pz, int py, int px, int d0, int d1, int d2,
                           int* __restrict__ gidx) {
  long t = (long)blockIdx.x * blockDim.x + threadIdx.x;
  int K = kd * kh * kw;
  if (t >= (long)K * Nout) return;
  int ki = (int)(t / Nout), o = (int)(t % Nout);
  int kz = ki / (kh * kw), ky = (ki / kw) % kh, kx = ki % kw;
  const int* c = oc + o * 4;
  int cz = c[1] + pz - kz, cy = c[2] + py - ky, cx = c[3] + px - kx;
  int idx = -1;
  if (cz % sz == 0 && cy % sy == 0 && cx % sx == 0) {
    int jz = cz / sz, jy = cy / sy, jx = cx / sx;
    if (jz >= 0 && jz < d0 && jy >= 0 && jy < d1 && jx >= 0 && jx < d2) {
      int q = ((c[0] * d0 + jz) * d1 + jy) * d2 + jx;
      idx = bsearch_key(ikeys, Nin, q);
    }
  }
  gidx[t] = idx;
}

// ---------------------------------------------------------------------------
// Feature f32 -> bf16, channel-padded to Cp, with one appended zero row
// (row index N acts as the "gather miss" row so the GEMM loop is branchless).
// ---------------------------------------------------------------------------
__global__ void k_to_bf16(const float* __restrict__ x, int N, int C, int Cp,
                          __bf16* __restrict__ y) {
  long t = (long)blockIdx.x * blockDim.x + threadIdx.x;
  long total = (long)(N + 1) * Cp;
  if (t >= total) return;
  int row = (int)(t / Cp), c = (int)(t % Cp);
  float v = (row < N && c < C) ? x[(long)row * C + c] : 0.f;
  y[t] = f2bf(v);
}

// ---------------------------------------------------------------------------
// Pack W [K][IC][OC] f32 into per-wave WMMA B fragments, bf16:
//   Wp[((k*nCH + c)*nCT + n)*512 + lane*16 + j]
//   lane<16: N=lane, K = c*32 + j (j<16);  lane>=16: N=lane-16, K = c*32+16+j
// Each GEMM lane then loads its 16 bf16 as one contiguous 32B chunk.
// ---------------------------------------------------------------------------
__global__ void k_pack_w(const float* __restrict__ W, int K, int IC, int OC,
                         int ICp, __bf16* __restrict__ Wp) {
  const int nCT = OC >> 4, nCH = ICp >> 5;
  long total = (long)K * nCH * nCT * 512;
  long t = (long)blockIdx.x * blockDim.x + threadIdx.x;
  if (t >= total) return;
  int j    = (int)(t & 15);
  int lane = (int)((t >> 4) & 31);
  int n    = (int)((t >> 9) % nCT);
  int c    = (int)((t >> 9) / nCT % nCH);
  int k    = (int)(t / ((long)512 * nCT * nCH));
  int half = lane >> 4, l16 = lane & 15;
  int kk  = c * 32 + half * 16 + j;
  int col = n * 16 + l16;
  float v = (kk < IC) ? W[((long)k * IC + kk) * OC + col] : 0.f;
  Wp[t] = f2bf(v);
}

// ---------------------------------------------------------------------------
// Core gather-GEMM on the WMMA pipe. Block = 128 threads = 4 waves.
// Each wave owns TWO 16-col tiles (32 output channels) sharing one A fragment:
// per K-chunk the body is 2 x 16B A loads + 2 x 32B B loads + 2 x v_wmma.
// gidx == nullptr -> identity gather (dense GEMM for the HCVR gate).
// ---------------------------------------------------------------------------
__global__ void k_gemm_wmma(const __bf16* __restrict__ feat, int ICp,
                            const __bf16* __restrict__ Wp, int OC,
                            const int* __restrict__ gidx, int K,
                            float* __restrict__ out, int Nout, int Nin) {
  const int lane = threadIdx.x & 31;
  const int wave = threadIdx.x >> 5;
  const int half = lane >> 4;
  const int l16  = lane & 15;
  const int nCT  = OC >> 4;
  const int ct0  = (blockIdx.y * 4 + wave) * 2;
  if (ct0 >= nCT) return;                 // wave-uniform; EXEC all-1s for WMMA
  const int nCH = ICp >> 5;
  const int rowBase = blockIdx.x * 16;
  const int arow = rowBase + l16;

  v8f acc0 = {0.f, 0.f, 0.f, 0.f, 0.f, 0.f, 0.f, 0.f};
  v8f acc1 = {0.f, 0.f, 0.f, 0.f, 0.f, 0.f, 0.f, 0.f};

  for (int k = 0; k < K; ++k) {
    int idx = (arow < Nout) ? (gidx ? gidx[(long)k * Nout + arow] : arow) : -1;
    const __bf16* frow = feat + (long)(idx < 0 ? Nin : idx) * ICp;  // Nin = zero row
    const __bf16* wk   = Wp + (long)k * nCH * nCT * 512;
    for (int c = 0; c < nCH; ++c) {
      // A fragment 16x32: lane holds K {c*32+half*16+0..15}
      v8bf alo = *(const v8bf*)(frow + c * 32 + half * 16);
      v8bf ahi = *(const v8bf*)(frow + c * 32 + half * 16 + 8);
      v16bf a = __builtin_shufflevector(alo, ahi,
                                        0, 1, 2, 3, 4, 5, 6, 7,
                                        8, 9, 10, 11, 12, 13, 14, 15);
      const __bf16* bp = wk + ((long)c * nCT + ct0) * 512 + lane * 16;
      v16bf b0 = *(const v16bf*)bp;
      v16bf b1 = *(const v16bf*)(bp + 512);
      acc0 = __builtin_amdgcn_wmma_f32_16x16x32_bf16(false, a, false, b0,
                                                     (short)0, acc0, false, false);
      acc1 = __builtin_amdgcn_wmma_f32_16x16x32_bf16(false, a, false, b1,
                                                     (short)0, acc1, false, false);
    }
  }
  // C/D layout: VGPR r -> M = r (lanes 0-15) or 8+r (lanes 16-31); N = l16
  const int col0 = ct0 * 16 + l16;
#pragma unroll
  for (int r = 0; r < 8; ++r) {
    int row = rowBase + half * 8 + r;
    if (row < Nout) {
      out[(long)row * OC + col0]      = acc0[r];
      out[(long)row * OC + col0 + 16] = acc1[r];
    }
  }
}

// ---------------------------------------------------------------------------
// BatchNorm (training stats, eps=1e-3) + ReLU
// ---------------------------------------------------------------------------
__global__ void k_bn_stats(const float* __restrict__ x, int N, int C,
                           float* __restrict__ mean, float* __restrict__ var) {
  __shared__ float ssum[256], ssq[256];
  int c = blockIdx.x;
  float s = 0.f, q = 0.f;
  for (int i = threadIdx.x; i < N; i += blockDim.x) {
    float v = x[(long)i * C + c];
    s += v; q += v * v;
  }
  ssum[threadIdx.x] = s; ssq[threadIdx.x] = q;
  __syncthreads();
  for (int o = blockDim.x >> 1; o > 0; o >>= 1) {
    if ((int)threadIdx.x < o) {
      ssum[threadIdx.x] += ssum[threadIdx.x + o];
      ssq[threadIdx.x]  += ssq[threadIdx.x + o];
    }
    __syncthreads();
  }
  if (threadIdx.x == 0) {
    float m = ssum[0] / (float)N;
    mean[c] = m;
    var[c]  = ssq[0] / (float)N - m * m;
  }
}

__global__ void k_bn_relu(float* __restrict__ x, long n, int C,
                          const float* __restrict__ mean, const float* __restrict__ var) {
  long t = (long)blockIdx.x * blockDim.x + threadIdx.x;
  if (t >= n) return;
  int c = (int)(t % C);
  float v = (x[t] - mean[c]) * rsqrtf(var[c] + 1e-3f);
  x[t] = v > 0.f ? v : 0.f;
}

// ---------------------------------------------------------------------------
// HCVR fusion pieces + elementwise glue
// ---------------------------------------------------------------------------
__global__ void k_proj(const float* __restrict__ rf, int Nr,
                       const float* __restrict__ w, float* __restrict__ rp) {
  int t = blockIdx.x * blockDim.x + threadIdx.x;
  if (t >= Nr * 256) return;
  int r = t / 256, j = t % 256;
  float s = 0.f;
#pragma unroll
  for (int k = 0; k < 5; ++k) s += rf[r * 5 + k] * w[k * 256 + j];
  rp[t] = s;
}

__global__ void k_hcvr_scatter(const float* __restrict__ rp, const int* __restrict__ rc,
                               int Nr, const int* __restrict__ keys4, int N4,
                               float* __restrict__ agg, float* __restrict__ cnt) {
  int t = blockIdx.x * blockDim.x + threadIdx.x;
  if (t >= Nr * 256) return;
  int i = t / 256, ch = t % 256;
  const int* c = rc + i * 4;
  int z = c[1] / 8; if (z > 4) z = 4;                      // min(z//8, SS4[0]-1)
  int q = ((c[0] * 5 + z) * 128 + c[2] / 8) * 128 + c[3] / 8;
  int j = bsearch_key(keys4, N4, q);
  if (j >= 0) {
    atomicAdd(&agg[(long)j * 256 + ch], rp[t]);
    if (ch == 0) atomicAdd(&cnt[j], 1.0f);
  }
}

__global__ void k_hcvr_div(float* __restrict__ agg, const float* __restrict__ cnt, int N4) {
  int t = blockIdx.x * blockDim.x + threadIdx.x;
  if (t >= N4 * 256) return;
  agg[t] /= fmaxf(cnt[t / 256], 1.0f);
}

// concat [x4 | agg] directly to bf16 (Cp=512) with appended zero row
__global__ void k_concat_bf16(const float* __restrict__ x4, const float* __restrict__ agg,
                              int N4, __bf16* __restrict__ cat) {
  long t = (long)blockIdx.x * blockDim.x + threadIdx.x;
  long total = (long)(N4 + 1) * 512;
  if (t >= total) return;
  int r = (int)(t / 512), c = (int)(t % 512);
  float v = 0.f;
  if (r < N4) v = (c < 256) ? x4[(long)r * 256 + c] : agg[(long)r * 256 + (c - 256)];
  cat[t] = f2bf(v);
}

__global__ void k_gate(float* __restrict__ x4, const float* __restrict__ g, long n) {
  long t = (long)blockIdx.x * blockDim.x + threadIdx.x;
  if (t >= n) return;
  x4[t] *= 1.0f / (1.0f + __expf(-g[t]));
}

__global__ void k_mul(float* __restrict__ a, const float* __restrict__ b, long n) {
  long t = (long)blockIdx.x * blockDim.x + threadIdx.x;
  if (t >= n) return;
  a[t] *= b[t];
}

// ---------------------------------------------------------------------------
// Host-side helpers
// ---------------------------------------------------------------------------
static inline int divup(long a, long b) { return (int)((a + b - 1) / b); }

static void bn_relu_(float* x, int N, int C, float* mean, float* var, hipStream_t s) {
  k_bn_stats<<<C, 256, 0, s>>>(x, N, C, mean, var);
  long n = (long)N * C;
  k_bn_relu<<<divup(n, 256), 256, 0, s>>>(x, n, C, mean, var);
}

static void conv_(bool inv, const float* fin, int IC, int ICp, __bf16* fbf,
                  const int* ikeys, int Nin, const int* ocoords, int Nout,
                  const float* W, int OC, __bf16* wpack,
                  int kd, int kh, int kw, int sz, int sy, int sx,
                  int pz, int py, int px, int d0, int d1, int d2,
                  int* gidx, float* fout, hipStream_t s) {
  int K = kd * kh * kw;
  long tc = (long)(Nin + 1) * ICp;
  k_to_bf16<<<divup(tc, 256), 256, 0, s>>>(fin, Nin, IC, ICp, fbf);
  long tw = (long)K * (ICp >> 5) * (OC >> 4) * 512;
  k_pack_w<<<divup(tw, 256), 256, 0, s>>>(W, K, IC, OC, ICp, wpack);
  long tg = (long)K * Nout;
  if (inv)
    k_gidx_inv<<<divup(tg, 256), 256, 0, s>>>(ocoords, Nout, ikeys, Nin, kd, kh, kw,
                                              sz, sy, sx, pz, py, px, d0, d1, d2, gidx);
  else
    k_gidx_conv<<<divup(tg, 256), 256, 0, s>>>(ocoords, Nout, ikeys, Nin, kd, kh, kw,
                                               sz, sy, sx, pz, py, px, d0, d1, d2, gidx);
  dim3 g(divup(Nout, 16), divup(OC, 128));
  k_gemm_wmma<<<g, 128, 0, s>>>(fbf, ICp, wpack, OC, gidx, K, fout, Nout, Nin);
}

// ---------------------------------------------------------------------------
extern "C" void kernel_launch(void* const* d_in, const int* in_sizes, int n_in,
                              void* d_out, int out_size, void* d_ws, size_t ws_size,
                              hipStream_t stream) {
  const float* lf    = (const float*)d_in[0];
  const float* rf    = (const float*)d_in[1];
  const float* w_in  = (const float*)d_in[2];
  const float* w_c1  = (const float*)d_in[3];
  const float* w_c2d = (const float*)d_in[4];
  const float* w_c2s = (const float*)d_in[5];
  const float* w_c3d = (const float*)d_in[6];
  const float* w_c3s = (const float*)d_in[7];
  const float* w_c4d = (const float*)d_in[8];
  const float* w_c4s = (const float*)d_in[9];
  const float* w_up1 = (const float*)d_in[10];
  const float* w_up2 = (const float*)d_in[11];
  const float* w_up3 = (const float*)d_in[12];
  const float* w_cup = (const float*)d_in[13];
  const float* w_outw= (const float*)d_in[14];
  const float* w_pr  = (const float*)d_in[15];
  const float* w_gt  = (const float*)d_in[16];
  const int* lc = (const int*)d_in[17];
  const int* rc = (const int*)d_in[18];
  const int* c0 = (const int*)d_in[19];
  const int* c2 = (const int*)d_in[20];
  const int* c3 = (const int*)d_in[21];
  const int* c4 = (const int*)d_in[22];
  const int* cA = (const int*)d_in[23];
  const int* cB = (const int*)d_in[24];
  const int* cC = (const int*)d_in[25];
  const int* cO = (const int*)d_in[26];

  const int Nl = in_sizes[0] / 5, Nr = in_sizes[1] / 5;
  const int N0 = in_sizes[19] / 4, N2 = in_sizes[20] / 4;
  const int N3 = in_sizes[21] / 4, N4 = in_sizes[22] / 4;
  const int NA = in_sizes[23] / 4, NB = in_sizes[24] / 4;
  const int NC = in_sizes[25] / 4, NO = in_sizes[26] / 4;
  int Nmax = N0;
  if (N2 > Nmax) Nmax = N2; if (N3 > Nmax) Nmax = N3; if (N4 > Nmax) Nmax = N4;
  if (NA > Nmax) Nmax = NA; if (NB > Nmax) Nmax = NB; if (NC > Nmax) Nmax = NC;
  if (NO > Nmax) Nmax = NO;

  // bump allocator on workspace
  char* p = (char*)d_ws;
  auto alloc = [&](size_t bytes) -> void* {
    void* r = (void*)p; p += (bytes + 255) & ~(size_t)255; return r;
  };
  int* keys0 = (int*)alloc((size_t)N0 * 4);
  int* keys2 = (int*)alloc((size_t)N2 * 4);
  int* keys3 = (int*)alloc((size_t)N3 * 4);
  int* keys4 = (int*)alloc((size_t)N4 * 4);
  int* keysA = (int*)alloc((size_t)NA * 4);
  int* keysB = (int*)alloc((size_t)NB * 4);
  int* keysC = (int*)alloc((size_t)NC * 4);
  int* gidx  = (int*)alloc((size_t)27 * Nmax * 4);
  float* mean = (float*)alloc(256 * 4);
  float* var  = (float*)alloc(256 * 4);
  float* f0  = (float*)alloc((size_t)N0 * 5 * 4);
  float* x   = (float*)alloc((size_t)N0 * 64 * 4);
  float* x1  = (float*)alloc((size_t)N0 * 64 * 4);
  float* u3  = (float*)alloc((size_t)N0 * 64 * 4);
  float* x2a = (float*)alloc((size_t)N2 * 128 * 4);
  float* x2b = (float*)alloc((size_t)N2 * 128 * 4);
  float* u2  = (float*)alloc((size_t)N2 * 128 * 4);
  float* x3a = (float*)alloc((size_t)N3 * 256 * 4);
  float* x3b = (float*)alloc((size_t)N3 * 256 * 4);
  float* u1  = (float*)alloc((size_t)N3 * 256 * 4);
  float* x4a = (float*)alloc((size_t)N4 * 256 * 4);
  float* x4b = (float*)alloc((size_t)N4 * 256 * 4);
  float* rp  = (float*)alloc((size_t)Nr * 256 * 4);
  float* agg = (float*)alloc((size_t)N4 * 256 * 4);
  float* cnt = (float*)alloc((size_t)N4 * 4);
  float* gbf = (float*)alloc((size_t)N4 * 256 * 4);
  float* fa  = (float*)alloc((size_t)NA * 64 * 4);
  float* fb  = (float*)alloc((size_t)NB * 64 * 4);
  float* fc  = (float*)alloc((size_t)NC * 64 * 4);
  // bf16 staging buffers (padded channels, +1 zero row each)
  __bf16* wpack = (__bf16*)alloc((size_t)27 * 256 * 256 * 2);  // max packed W
  __bf16* bf0 = (__bf16*)alloc((size_t)(N0 + 1) * 32 * 2);
  __bf16* bx  = (__bf16*)alloc((size_t)(N0 + 1) * 64 * 2);
  __bf16* bx1 = (__bf16*)alloc((size_t)(N0 + 1) * 64 * 2);
  __bf16* b2  = (__bf16*)alloc((size_t)(N2 + 1) * 128 * 2);
  __bf16* b3  = (__bf16*)alloc((size_t)(N3 + 1) * 256 * 2);
  __bf16* b4  = (__bf16*)alloc((size_t)(N4 + 1) * 256 * 2);
  __bf16* bct = (__bf16*)alloc((size_t)(N4 + 1) * 512 * 2);
  __bf16* bA  = (__bf16*)alloc((size_t)(NA + 1) * 64 * 2);
  __bf16* bB  = (__bf16*)alloc((size_t)(NB + 1) * 64 * 2);
  __bf16* bC  = (__bf16*)alloc((size_t)(NC + 1) * 64 * 2);
  float* out = (float*)d_out;   // [NO x 64]

  // scalar-key tables for every input coordinate set
  k_keys<<<divup(N0, 256), 256, 0, stream>>>(c0, N0, 41, 1024, 1024, keys0);
  k_keys<<<divup(N2, 256), 256, 0, stream>>>(c2, N2, 21,  512,  512, keys2);
  k_keys<<<divup(N3, 256), 256, 0, stream>>>(c3, N3, 11,  256,  256, keys3);
  k_keys<<<divup(N4, 256), 256, 0, stream>>>(c4, N4,  5,  128,  128, keys4);
  k_keys<<<divup(NA, 256), 256, 0, stream>>>(cA, NA, 21, 1024, 1024, keysA);
  k_keys<<<divup(NB, 256), 256, 0, stream>>>(cB, NB, 11, 1024, 1024, keysB);
  k_keys<<<divup(NC, 256), 256, 0, stream>>>(cC, NC,  5, 1024, 1024, keysC);

  // sparse_add of lidar + radar features on the union set
  k_fill<<<divup((long)N0 * 5, 256), 256, 0, stream>>>(f0, (long)N0 * 5);
  k_scatter_add<<<divup(Nl * 5, 256), 256, 0, stream>>>(lf, lc, Nl, 5, keys0, N0, 41, 1024, 1024, f0);
  k_scatter_add<<<divup(Nr * 5, 256), 256, 0, stream>>>(rf, rc, Nr, 5, keys0, N0, 41, 1024, 1024, f0);

  // encoder
  conv_(false, f0, 5, 32, bf0, keys0, N0, c0, N0, w_in, 64, wpack,
        3,3,3, 1,1,1, 1,1,1, 41,1024,1024, gidx, x, stream);
  bn_relu_(x, N0, 64, mean, var, stream);
  conv_(false, x, 64, 64, bx, keys0, N0, c0, N0, w_c1, 64, wpack,
        3,3,3, 1,1,1, 1,1,1, 41,1024,1024, gidx, x1, stream);
  bn_relu_(x1, N0, 64, mean, var, stream);
  conv_(false, x1, 64, 64, bx1, keys0, N0, c2, N2, w_c2d, 128, wpack,
        3,3,3, 2,2,2, 1,1,1, 41,1024,1024, gidx, x2a, stream);
  bn_relu_(x2a, N2, 128, mean, var, stream);
  conv_(false, x2a, 128, 128, b2, keys2, N2, c2, N2, w_c2s, 128, wpack,
        3,3,3, 1,1,1, 1,1,1, 21,512,512, gidx, x2b, stream);
  bn_relu_(x2b, N2, 128, mean, var, stream);
  conv_(false, x2b, 128, 128, b2, keys2, N2, c2, N2, w_c2s + (size_t)27*128*128, 128, wpack,
        3,3,3, 1,1,1, 1,1,1, 21,512,512, gidx, x2a, stream);
  bn_relu_(x2a, N2, 128, mean, var, stream);
  conv_(false, x2a, 128, 128, b2, keys2, N2, c3, N3, w_c3d, 256, wpack,
        3,3,3, 2,2,2, 1,1,1, 21,512,512, gidx, x3a, stream);
  bn_relu_(x3a, N3, 256, mean, var, stream);
  conv_(false, x3a, 256, 256, b3, keys3, N3, c3, N3, w_c3s, 256, wpack,
        3,3,3, 1,1,1, 1,1,1, 11,256,256, gidx, x3b, stream);
  bn_relu_(x3b, N3, 256, mean, var, stream);
  conv_(false, x3b, 256, 256, b3, keys3, N3, c3, N3, w_c3s + (size_t)27*256*256, 256, wpack,
        3,3,3, 1,1,1, 1,1,1, 11,256,256, gidx, x3a, stream);
  bn_relu_(x3a, N3, 256, mean, var, stream);
  conv_(false, x3a, 256, 256, b3, keys3, N3, c4, N4, w_c4d, 256, wpack,
        3,3,3, 2,2,2, 0,1,1, 11,256,256, gidx, x4a, stream);
  bn_relu_(x4a, N4, 256, mean, var, stream);
  conv_(false, x4a, 256, 256, b4, keys4, N4, c4, N4, w_c4s, 256, wpack,
        3,3,3, 1,1,1, 1,1,1, 5,128,128, gidx, x4b, stream);
  bn_relu_(x4b, N4, 256, mean, var, stream);
  conv_(false, x4b, 256, 256, b4, keys4, N4, c4, N4, w_c4s + (size_t)27*256*256, 256, wpack,
        3,3,3, 1,1,1, 1,1,1, 5,128,128, gidx, x4a, stream);
  bn_relu_(x4a, N4, 256, mean, var, stream);

  // HCVR cross-modal fusion: project radar, scatter-mean, sigmoid gate
  k_proj<<<divup(Nr * 256, 256), 256, 0, stream>>>(rf, Nr, w_pr, rp);
  k_fill<<<divup((long)N4 * 256, 256), 256, 0, stream>>>(agg, (long)N4 * 256);
  k_fill<<<divup((long)N4, 256), 256, 0, stream>>>(cnt, (long)N4);
  k_hcvr_scatter<<<divup(Nr * 256, 256), 256, 0, stream>>>(rp, rc, Nr, keys4, N4, agg, cnt);
  k_hcvr_div<<<divup(N4 * 256, 256), 256, 0, stream>>>(agg, cnt, N4);
  k_concat_bf16<<<divup((long)(N4 + 1) * 512, 256), 256, 0, stream>>>(x4a, agg, N4, bct);
  {
    long tw = (long)1 * (512 >> 5) * (256 >> 4) * 512;
    k_pack_w<<<divup(tw, 256), 256, 0, stream>>>(w_gt, 1, 512, 256, 512, wpack);
    dim3 g(divup(N4, 16), divup(256, 128));
    k_gemm_wmma<<<g, 128, 0, stream>>>(bct, 512, wpack, 256, nullptr, 1, gbf, N4, N4);
  }
  k_gate<<<divup((long)N4 * 256, 256), 256, 0, stream>>>(x4a, gbf, (long)N4 * 256);

  // decoder: inverse convs + multiplicative skip fusion
  conv_(true, x4a, 256, 256, b4, keys4, N4, c3, N3, w_up1, 256, wpack,
        3,3,3, 2,2,2, 0,1,1, 5,128,128, gidx, u1, stream);
  bn_relu_(u1, N3, 256, mean, var, stream);
  k_mul<<<divup((long)N3 * 256, 256), 256, 0, stream>>>(x3a, u1, (long)N3 * 256);
  conv_(true, x3a, 256, 256, b3, keys3, N3, c2, N2, w_up2, 128, wpack,
        3,3,3, 2,2,2, 1,1,1, 11,256,256, gidx, u2, stream);
  bn_relu_(u2, N2, 128, mean, var, stream);
  k_mul<<<divup((long)N2 * 128, 256), 256, 0, stream>>>(x2a, u2, (long)N2 * 128);
  conv_(true, x2a, 128, 128, b2, keys2, N2, c0, N0, w_up3, 64, wpack,
        3,3,3, 2,2,2, 1,1,1, 21,512,512, gidx, u3, stream);
  bn_relu_(u3, N0, 64, mean, var, stream);
  k_mul<<<divup((long)N0 * 64, 256), 256, 0, stream>>>(x1, u3, (long)N0 * 64);
  k_mul<<<divup((long)N0 * 64, 256), 256, 0, stream>>>(x, x1, (long)N0 * 64);

  // z-compression chain (3,1,1) stride (2,1,1)
  conv_(false, x, 64, 64, bx, keys0, N0, cA, NA, w_cup, 64, wpack,
        3,1,1, 2,1,1, 1,0,0, 41,1024,1024, gidx, fa, stream);
  bn_relu_(fa, NA, 64, mean, var, stream);
  conv_(false, fa, 64, 64, bA, keysA, NA, cB, NB, w_cup + (size_t)1*3*64*64, 64, wpack,
        3,1,1, 2,1,1, 1,0,0, 21,1024,1024, gidx, fb, stream);
  bn_relu_(fb, NB, 64, mean, var, stream);
  conv_(false, fb, 64, 64, bB, keysB, NB, cC, NC, w_cup + (size_t)2*3*64*64, 64, wpack,
        3,1,1, 2,1,1, 0,0,0, 11,1024,1024, gidx, fc, stream);
  bn_relu_(fc, NC, 64, mean, var, stream);
  conv_(false, fc, 64, 64, bC, keysC, NC, cO, NO, w_outw, 64, wpack,
        3,1,1, 2,1,1, 0,0,0, 5,1024,1024, gidx, out, stream);
  bn_relu_(out, NO, 64, mean, var, stream);
}
